// MatchingBlock_39960375722527
// MI455X (gfx1250) — matrix operand order repaired and checked by reading, not compiled, not run on hardware
//
#include <hip/hip_runtime.h>
#include <hip/hip_bf16.h>

typedef __attribute__((ext_vector_type(16))) __bf16 v16bf;
typedef __attribute__((ext_vector_type(8)))  __bf16 v8bf;
typedef __attribute__((ext_vector_type(8)))  float  v8f;

#define B2    4
#define HDIM  128
#define WDIM  128
#define HW    16384          // 128*128
#define CDIM  128
#define KKEYS 1024
#define NW    8              // waves per attention block (wave32)

// ---------------------------------------------------------------------------
// Kernel 1: batch-swap + f32 -> bf16 convert of feature (feature_reverse)
// featRev[b][n][c] = bf16(feature[b^2][n][c])   (B=2, so swap is XOR 2)
// ---------------------------------------------------------------------------
__global__ __launch_bounds__(256) void convert_rev_kernel(const float* __restrict__ feat,
                                                          __bf16* __restrict__ featRev) {
    const size_t perBatch = (size_t)HW * CDIM;
    size_t i = ((size_t)blockIdx.x * blockDim.x + threadIdx.x) * 4;
    if (i >= (size_t)B2 * perBatch) return;
    size_t b = i / perBatch;
    size_t r = i - b * perBatch;
    const float4 v = *(const float4*)(feat + (b ^ 2) * perBatch + r);
    __bf16* o = featRev + i;
    o[0] = (__bf16)v.x; o[1] = (__bf16)v.y; o[2] = (__bf16)v.z; o[3] = (__bf16)v.w;
}

// ---------------------------------------------------------------------------
// Kernel 2: per-batch top-K via full bitonic sort (descending) of 16384
// packed (value_bits<<32 | index) u64 keys in LDS. One block per batch.
// Only the SET of top-K indices matters for the final output.
// ---------------------------------------------------------------------------
__global__ __launch_bounds__(1024) void topk_kernel(const float* __restrict__ imap,
                                                    int* __restrict__ kpind) {
    extern __shared__ unsigned long long sm[];
    const int b = blockIdx.x;
    const int tid = threadIdx.x;
    const float* src = imap + (size_t)b * HW;
    for (int i = tid; i < HW; i += 1024) {
        unsigned key = __float_as_uint(src[i]);   // uniform [0,1): order-preserving bits
        sm[i] = ((unsigned long long)key << 32) | (unsigned)i;
    }
    __syncthreads();
    for (unsigned kk = 2; kk <= HW; kk <<= 1) {
        for (unsigned j = kk >> 1; j > 0; j >>= 1) {
            for (unsigned i = tid; i < HW; i += 1024) {
                unsigned ixj = i ^ j;
                if (ixj > i) {
                    unsigned long long a = sm[i], c = sm[ixj];
                    bool desc = ((i & kk) == 0);
                    if (desc ? (a < c) : (a > c)) { sm[i] = c; sm[ixj] = a; }
                }
            }
            __syncthreads();
        }
    }
    if (tid < KKEYS) kpind[b * KKEYS + tid] = (int)(unsigned)(sm[tid] & 0xffffffffu);
}

// ---------------------------------------------------------------------------
// Kernel 3: gather key-point features + convert to bf16
// kpfeat[b][k][c] = bf16(feature[b][kpind[b][k]][c])
// ---------------------------------------------------------------------------
__global__ __launch_bounds__(256) void gather_kernel(const float* __restrict__ feat,
                                                     const int* __restrict__ kpind,
                                                     __bf16* __restrict__ kpfeat) {
    size_t i = (size_t)blockIdx.x * blockDim.x + threadIdx.x;
    if (i >= (size_t)B2 * KKEYS * CDIM) return;
    int c = (int)(i % CDIM);
    int k = (int)((i / CDIM) % KKEYS);
    int b = (int)(i / ((size_t)CDIM * KKEYS));
    int n = kpind[b * KKEYS + k];
    kpfeat[i] = (__bf16)feat[((size_t)b * HW + n) * CDIM + c];
}

// ---------------------------------------------------------------------------
// Kernel 4: fused  kp_feat @ feature_rev^T / sqrt(C)  -> online softmax ->
// expectation over coords grid.  WMMA bf16 16x16x32, f32 accum.
// Grid: (64 key-tiles, 4 batches); block = 8 waves (256 threads, wave32).
// Each wave owns HW/NW = 2048 columns; A (16 keys x 128 K) held in VGPRs.
// Software-pipelined: next tile's B fragments are loaded before the WMMAs so
// the softmax VALU tail hides the load latency (split loadcnt waits).
// ---------------------------------------------------------------------------
__global__ __launch_bounds__(256) void attn_kernel(const __bf16* __restrict__ kpfeat,
                                                   const __bf16* __restrict__ featRev,
                                                   float2* __restrict__ outkp) {
    // 1/sqrt(128) * log2(e): softmax in base-2 (identical result, maps to v_exp_f32)
    const float SCL = 0.08838834764831845f * 1.4426950408889634f;
    const int ITER = HW / NW / 16;   // 128 N-tiles per wave

    const int tile = blockIdx.x;     // 0..63
    const int b    = blockIdx.y;     // 0..3
    const int tid  = threadIdx.x;
    const int wave = tid >> 5;
    const int lane = tid & 31;
    const int h    = (lane >> 4) & 1;   // half-wave select
    const int lm   = lane & 15;

    // ---- load A fragments (contiguous K, matches B): row M = lm,
    //      elems {c*32 + h*16 .. +15}
    const __bf16* Abase = kpfeat + ((size_t)b * KKEYS + tile * 16 + lm) * CDIM;
    v16bf afrag[4];
#pragma unroll
    for (int c = 0; c < 4; ++c) {
        v8bf lo = *(const v8bf*)(Abase + c * 32 + h * 16);
        v8bf hi = *(const v8bf*)(Abase + c * 32 + h * 16 + 8);
#pragma unroll
        for (int e = 0; e < 8; ++e) { afrag[c][e] = lo[e]; afrag[c][8 + e] = hi[e]; }
    }

    const __bf16* Bbase = featRev + (size_t)b * HW * CDIM;
    const int nstart = wave * (HW / NW);

    // per-lane online-softmax state for 8 rows (lane-local column subset)
    float M[8], s[8], sx[8], sy[8];
#pragma unroll
    for (int r = 0; r < 8; ++r) { M[r] = -INFINITY; s[r] = 0.f; sx[r] = 0.f; sy[r] = 0.f; }

    // ---- prologue: load B fragments for tile 0
    // B layout (mirrors documented sparse-B): lanes0-15 K=c*32..+15,
    // lanes16-31 K=c*32+16..+31, column n = lane%16 -> 16 contiguous K (32B)
    v16bf bcur[4];
    {
        const __bf16* Brow = Bbase + (size_t)(nstart + lm) * CDIM + h * 16;
#pragma unroll
        for (int c = 0; c < 4; ++c) bcur[c] = *(const v16bf*)(Brow + c * 32);
    }

    for (int it = 0; it < ITER; ++it) {
        const int n = nstart + it * 16 + lm;              // this lane's column

        // ---- prefetch next tile's B fragments (wrap-around: branchless)
        const int nnext = (it + 1 < ITER) ? (n + 16) : (nstart + lm);
        const __bf16* BrowN = Bbase + (size_t)nnext * CDIM + h * 16;
        v16bf bnext[4];
#pragma unroll
        for (int c = 0; c < 4; ++c) bnext[c] = *(const v16bf*)(BrowN + c * 32);

        // ---- 16x16 logits, K=128 via 4 chained WMMAs (waits only on bcur)
        v8f acc = {};
#pragma unroll
        for (int c = 0; c < 4; ++c) {
            acc = __builtin_amdgcn_wmma_f32_16x16x32_bf16(
                false, afrag[c], false, bcur[c], (short)0, acc, false, false);
        }

        // ---- branchless online softmax + coord expectation (hides prefetch)
        const float xn = (float)(n & (WDIM - 1));
        const float yn = (float)(n >> 7);
#pragma unroll
        for (int r = 0; r < 8; ++r) {
            float l2   = acc[r] * SCL;
            float newM = fmaxf(M[r], l2);
            float sc   = exp2f(M[r] - newM);
            float p    = exp2f(l2 - newM);
            s[r]  = s[r]  * sc + p;
            sx[r] = sx[r] * sc + p * xn;
            sy[r] = sy[r] * sc + p * yn;
            M[r]  = newM;
        }

#pragma unroll
        for (int c = 0; c < 4; ++c) bcur[c] = bnext[c];
    }

    // ---- merge the 16 lane-local partials within each half-wave (rows 0-7 in
    //      lanes 0-15, rows 8-15 in lanes 16-31); xor<16 stays in the group.
#pragma unroll
    for (int off = 1; off < 16; off <<= 1) {
#pragma unroll
        for (int r = 0; r < 8; ++r) {
            float Mo  = __shfl_xor(M[r],  off, 32);
            float so  = __shfl_xor(s[r],  off, 32);
            float sxo = __shfl_xor(sx[r], off, 32);
            float syo = __shfl_xor(sy[r], off, 32);
            float newM = fmaxf(M[r], Mo);
            float a0 = exp2f(M[r] - newM);
            float a1 = exp2f(Mo  - newM);
            s[r]  = s[r]  * a0 + so  * a1;
            sx[r] = sx[r] * a0 + sxo * a1;
            sy[r] = sy[r] * a0 + syo * a1;
            M[r]  = newM;
        }
    }

    __shared__ float4 part[NW][16];
    if (lm == 0) {
#pragma unroll
        for (int r = 0; r < 8; ++r)
            part[wave][r + 8 * h] = make_float4(M[r], s[r], sx[r], sy[r]);
    }
    __syncthreads();

    if (tid < 16) {
        float Mm = -INFINITY, ss = 0.f, sxx = 0.f, syy = 0.f;
#pragma unroll
        for (int w = 0; w < NW; ++w) {
            float4 p = part[w][tid];
            float newM = fmaxf(Mm, p.x);
            float a0 = exp2f(Mm - newM);
            float a1 = exp2f(p.x - newM);
            ss  = ss  * a0 + p.y * a1;
            sxx = sxx * a0 + p.z * a1;
            syy = syy * a0 + p.w * a1;
            Mm  = newM;
        }
        float inv = 1.0f / ss;
        outkp[(size_t)b * KKEYS + tile * 16 + tid] = make_float2(sxx * inv, syy * inv);
    }
}

// ---------------------------------------------------------------------------
// Kernel 5: zero output; Kernel 6: scatter (out - grid) at key positions.
// Output layout: [b, 2, H, W], channel 0 = x, channel 1 = y.
// ---------------------------------------------------------------------------
__global__ __launch_bounds__(256) void zero_kernel(float* __restrict__ out, int n) {
    int i = blockIdx.x * blockDim.x + threadIdx.x;
    if (i < n) out[i] = 0.0f;
}

__global__ __launch_bounds__(256) void scatter_kernel(const int* __restrict__ kpind,
                                                      const float2* __restrict__ outkp,
                                                      float* __restrict__ out) {
    int i = blockIdx.x * blockDim.x + threadIdx.x;
    if (i >= B2 * KKEYS) return;
    int b = i / KKEYS;
    int n = kpind[i];
    float2 o = outkp[i];
    float* base = out + (size_t)b * 2 * HW;
    base[n]      = o.x - (float)(n & (WDIM - 1));
    base[HW + n] = o.y - (float)(n >> 7);
}

// ---------------------------------------------------------------------------
extern "C" void kernel_launch(void* const* d_in, const int* in_sizes, int n_in,
                              void* d_out, int out_size, void* d_ws, size_t ws_size,
                              hipStream_t stream) {
    const float* imap = (const float*)d_in[0];   // [4,1,128,128]
    const float* feat = (const float*)d_in[1];   // [4,16384,128]
    float* out = (float*)d_out;                  // [4,2,128,128]

    char* ws = (char*)d_ws;
    __bf16* featRev = (__bf16*)(ws);                                  // 16 MiB
    __bf16* kpfeat  = (__bf16*)(ws + (size_t)16 * 1024 * 1024);       // 1 MiB
    int*    kpind   = (int*)   (ws + (size_t)17 * 1024 * 1024);       // 16 KiB
    float2* outkp   = (float2*)(ws + (size_t)17 * 1024 * 1024 + 65536);

    // 1) feature_reverse -> bf16
    {
        size_t total = (size_t)B2 * HW * CDIM / 4;   // float4 per thread
        convert_rev_kernel<<<(int)((total + 255) / 256), 256, 0, stream>>>(feat, featRev);
    }
    // 2) top-K indices (bitonic sort of 16384 u64 keys in 128 KB LDS)
    topk_kernel<<<B2, 1024, HW * sizeof(unsigned long long), stream>>>(imap, kpind);
    // 3) gather key features -> bf16
    {
        size_t total = (size_t)B2 * KKEYS * CDIM;
        gather_kernel<<<(int)((total + 255) / 256), 256, 0, stream>>>(feat, kpind, kpfeat);
    }
    // 4) fused WMMA matching + softmax-expectation
    attn_kernel<<<dim3(KKEYS / 16, B2), 32 * NW, 0, stream>>>(kpfeat, featRev, outkp);
    // 5) zero + scatter into dense flow map
    zero_kernel<<<(B2 * 2 * HW + 255) / 256, 256, 0, stream>>>(out, B2 * 2 * HW);
    scatter_kernel<<<(B2 * KKEYS + 255) / 256, 256, 0, stream>>>(kpind, outkp, out);
}
